// MultiAxisAttention_3616362463275
// MI455X (gfx1250) — matrix operand order repaired and checked by reading, not compiled, not run on hardware
//
#include <hip/hip_runtime.h>
#include <hip/hip_bf16.h>

typedef __attribute__((ext_vector_type(16))) __bf16 v16bf;
typedef __attribute__((ext_vector_type(8)))  float  v8f;

#define N_   8
#define C_   64
#define CQ_  16
#define H_   256
#define W_   256
#define HW_  65536
#define NBLK 256

// workspace offsets (bytes)
#define WS_XEMBED 0
#define WS_WRA    33554432              // after x_embed (8*16*65536*4 B)
#define WS_WASMA  (33554432 + 18432)    // after wrA (18*32*16*2 B)

// ---- attention kernel LDS layout (bytes). Phase-based union:
#define XS_PITCH 40
#define PS_PITCH 40
#define VT_PITCH 272
#define OFF_QS 0
#define OFF_KS 8192
#define OFF_PS 16384
#define OFF_VT 26624
#define SMEM_BYTES (OFF_VT + C_ * VT_PITCH * 2)   // 61440 B

// ---------------------------------------------------------------------------
// One-time weight pre-pack: store bf16 A-fragments in exact WMMA lane layout.
//   wrA  [18][32][16]: conv3x3 weights, K-order k = tap*64 + ci
//   wasmA [8][32][16]: 1x1 conv weights, group g = kh*4 + mt
// Fragment element t: K' = half*16 + t
// ---------------------------------------------------------------------------
__global__ __launch_bounds__(256, 1)
void pack_weights_kernel(const float* __restrict__ wgt,   // [16,64,3,3]
                         const float* __restrict__ wasm,  // [64,64]
                         __bf16* __restrict__ wrA,
                         __bf16* __restrict__ wasmA)
{
    const int tid = threadIdx.x;
    // conv3x3 fragments: 18 K-steps x 32 lanes x 16 elements
    for (int idx = tid; idx < 18 * 32 * 16; idx += 256) {
        const int t    = idx & 15;
        const int lane = (idx >> 4) & 31;
        const int s    = idx >> 9;
        const int half = lane >> 4;
        const int l15  = lane & 15;
        const int tap  = s >> 1;
        const int cib  = (s & 1) * 32;
        const int kp   = half * 16 + t;
        const int ci   = cib + kp;
        wrA[idx] = (__bf16)wgt[(l15 * 64 + ci) * 9 + tap];
    }
    // 1x1 conv fragments: (kh*4+mt) x 32 lanes x 16 elements
    for (int idx = tid; idx < 8 * 32 * 16; idx += 256) {
        const int t    = idx & 15;
        const int lane = (idx >> 4) & 31;
        const int g    = idx >> 9;          // kh*4 + mt
        const int kh   = g >> 2;
        const int mt   = g & 3;
        const int half = lane >> 4;
        const int l15  = lane & 15;
        const int c    = mt * 16 + l15;
        const int kp   = half * 16 + t;
        wasmA[idx] = (__bf16)wasm[c * 64 + kh * 32 + kp];
    }
}

// ---------------------------------------------------------------------------
// 3x3 conv (x -> x_embed, Cq=16 out channels) as implicit-GEMM on WMMA.
//   M = 16 (cq), K = 576 ordered k = tap*64 + ci (tap-major), N = pixels.
//   Workgroup: 4 output rows x 64 cols for one n. LDS patch 6 x 66 x 64 (bf16).
// ---------------------------------------------------------------------------
#define CXP 72                      // ci pitch in LDS (64 + 8 pad vs banks)
#define CONV_SMEM (6 * 66 * CXP)    // 28512 bf16 = 57024 B

__global__ __launch_bounds__(256, 1)
void conv3x3_wmma_kernel(const float* __restrict__ x,     // [8,64,256,256]
                         const __bf16* __restrict__ wrA,  // packed A-frags
                         const float* __restrict__ bias,  // [16]
                         float* __restrict__ y)           // [8,16,256,256]
{
    __shared__ __bf16 Xh[CONV_SMEM];

    const int w0   = blockIdx.x * 64;
    const int h0   = blockIdx.y * 4;
    const int n    = blockIdx.z;
    const int tid  = threadIdx.x;
    const int wave = tid >> 5;
    const int lane = tid & 31;
    const int half = lane >> 4;
    const int l15  = lane & 15;

    // ---- stage input patch (rows h0-1 .. h0+4, cols w0-1 .. w0+64), zero-pad OOB
    for (int idx = tid; idx < 6 * 66 * 64; idx += 256) {
        const int wc   = idx % 66;
        const int rest = idx / 66;      // r*64 + ci
        const int ci   = rest & 63;
        const int r    = rest >> 6;
        const int h    = h0 + r - 1;
        const int w    = w0 + wc - 1;
        float v = 0.f;
        if (h >= 0 && h < H_ && w >= 0 && w < W_)
            v = x[(((size_t)n * C_ + ci) * H_ + h) * W_ + w];
        Xh[(r * 66 + wc) * CXP + ci] = (__bf16)v;
    }
    __syncthreads();

    // ---- 18 K-steps of 32; each wave owns 2 of the 16 N-tiles
    v8f acc[2];
    acc[0] = (v8f){0.f, 0.f, 0.f, 0.f, 0.f, 0.f, 0.f, 0.f};
    acc[1] = (v8f){0.f, 0.f, 0.f, 0.f, 0.f, 0.f, 0.f, 0.f};

    for (int s = 0; s < 18; ++s) {
        const int tap = s >> 1;
        const int dh3 = tap / 3;        // (dh+1)
        const int dw3 = tap % 3;        // (dw+1)
        const int cib = (s & 1) * 32;

        // A fragment: one aligned 32-byte load from the packed table
        const v16bf aW = *(const v16bf*)(wrA + ((s * 32 + lane) << 4));

        #pragma unroll
        for (int nt2 = 0; nt2 < 2; ++nt2) {
            const int nt   = wave * 2 + nt2;
            const int j    = nt * 16 + l15;    // pixel index: hrow*64 + wcol
            const int hrow = j >> 6;
            const int wcol = j & 63;
            const __bf16* bp =
                &Xh[((hrow + dh3) * 66 + (wcol + dw3)) * CXP + cib + half * 16];
            v16bf bX;
            #pragma unroll
            for (int t = 0; t < 16; ++t) bX[t] = bp[t];
            acc[nt2] = __builtin_amdgcn_wmma_f32_16x16x32_bf16(
                false, aW, false, bX, (short)0, acc[nt2], false, false);
        }
    }

    // ---- epilogue: D row = cq (e + half*8), col = pixel; coalesced stores
    #pragma unroll
    for (int nt2 = 0; nt2 < 2; ++nt2) {
        const int nt   = wave * 2 + nt2;
        const int hrow = nt >> 2;
        const int wcol = (nt & 3) * 16 + l15;
        #pragma unroll
        for (int e = 0; e < 8; ++e) {
            const int cq = e + half * 8;
            y[(((size_t)n * CQ_ + cq) * H_ + (h0 + hrow)) * W_ + (w0 + wcol)] =
                acc[nt2][e] + bias[cq];
        }
    }
}

// ---------------------------------------------------------------------------
// Per-(n,k) attention over blocks, with the 1x1 conv (V projection) fused in
// as a 64x64x256 bf16-WMMA GEMM on the gathered pixel set.
// ---------------------------------------------------------------------------
__global__ __launch_bounds__(256, 1)
void attn_kernel(const float* __restrict__ xe,     // x_embed [8,16,256,256]
                 const float* __restrict__ x,      // [8,64,256,256]
                 const __bf16* __restrict__ wasmA, // packed A-frags
                 const float* __restrict__ basm,   // [64]
                 float* __restrict__ out)          // [8,64,256,256]
{
    __shared__ __align__(16) unsigned char smem[SMEM_BYTES];
    __bf16 (*Xs)[XS_PITCH]     = (__bf16(*)[XS_PITCH])(smem);
    __bf16 (*Qs)[CQ_]          = (__bf16(*)[CQ_])(smem + OFF_QS);
    __bf16 (*Ks)[CQ_]          = (__bf16(*)[CQ_])(smem + OFF_KS);
    __bf16 (*Ps)[16][PS_PITCH] = (__bf16(*)[16][PS_PITCH])(smem + OFF_PS);
    __bf16 (*Vt)[VT_PITCH]     = (__bf16(*)[VT_PITCH])(smem + OFF_VT);

    const int wg   = blockIdx.x;      // n*256 + k
    const int n    = wg >> 8;
    const int k    = wg & 255;
    const int bh   = k >> 4;
    const int bw   = k & 15;
    const int tid  = threadIdx.x;
    const int wave = tid >> 5;
    const int lane = tid & 31;
    const int half = lane >> 4;
    const int l15  = lane & 15;

    // =======================================================================
    // Phase 0: Vt[c][j] = sum_ci wasm[c][ci] * x[n][ci][h(j)][w(j)] + basm[c]
    // =======================================================================
    v8f vacc[8];
    #pragma unroll
    for (int tt = 0; tt < 8; ++tt)
        vacc[tt] = (v8f){0.f, 0.f, 0.f, 0.f, 0.f, 0.f, 0.f, 0.f};

    for (int kh = 0; kh < 2; ++kh) {
        {
            const int j = tid;
            const int h = (j & 0xF0) + bh;
            const int w = ((j & 15) << 4) + bw;
            const size_t pix = (size_t)h * W_ + w;
            #pragma unroll 4
            for (int it = 0; it < 32; ++it) {
                const int c = kh * 32 + it;
                Xs[j][it] = (__bf16)x[((size_t)n * C_ + c) * (size_t)HW_ + pix];
            }
        }
        __syncthreads();

        #pragma unroll
        for (int tt = 0; tt < 8; ++tt) {
            const int tile = wave * 8 + tt;
            const int mt   = tile >> 4;
            const int nt   = tile & 15;
            const v16bf aW =
                *(const v16bf*)(wasmA + (((kh * 4 + mt) * 32 + lane) << 4));
            const __bf16* xp = &Xs[nt * 16 + l15][half * 16];
            v16bf bX;
            #pragma unroll
            for (int t = 0; t < 16; ++t) bX[t] = xp[t];
            vacc[tt] = __builtin_amdgcn_wmma_f32_16x16x32_bf16(
                false, aW, false, bX, (short)0, vacc[tt], false, false);
        }
        __syncthreads();
    }
    #pragma unroll
    for (int tt = 0; tt < 8; ++tt) {
        const int tile = wave * 8 + tt;
        const int mt   = tile >> 4;
        const int nt   = tile & 15;
        const int j    = nt * 16 + l15;
        #pragma unroll
        for (int e = 0; e < 8; ++e) {
            const int c = mt * 16 + e + half * 8;
            Vt[c][j] = (__bf16)(vacc[tt][e] + basm[c]);
        }
    }

    // =======================================================================
    // Phase 1: load Q rows from x_embed, L2-normalize -> K (bf16 in LDS).
    // =======================================================================
    {
        const int j = tid;
        const int h = (j & 0xF0) + bh;
        const int w = ((j & 15) << 4) + bw;
        const size_t base = (size_t)n * CQ_ * (size_t)HW_ + (size_t)h * W_ + w;
        float q[CQ_];
        float ss = 0.f;
        #pragma unroll
        for (int e = 0; e < CQ_; ++e) {
            q[e] = xe[base + (size_t)e * HW_];
            ss += q[e] * q[e];
        }
        const float inv = 1.0f / fmaxf(sqrtf(ss), 5e-5f);
        #pragma unroll
        for (int e = 0; e < CQ_; ++e) {
            Qs[j][e] = (__bf16)q[e];
            Ks[j][e] = (__bf16)(q[e] * inv);
        }
    }
    __syncthreads();

    // =======================================================================
    // Phase 2: attention. Each wave handles two 16-row strips.
    // =======================================================================
    for (int s = 0; s < 2; ++s) {
        const int strip = wave + s * 8;

        v16bf aQ;
        {
            const __bf16* qp = &Qs[strip * 16 + l15][half * 8];
            #pragma unroll
            for (int t = 0; t < 8; ++t) aQ[t] = qp[t];
            #pragma unroll
            for (int t = 8; t < 16; ++t) aQ[t] = (__bf16)0.0f;
        }

        // ---- pass 1: row maxima over all 256 columns
        float mx[8];
        #pragma unroll
        for (int e = 0; e < 8; ++e) mx[e] = -1e30f;

        for (int jt = 0; jt < 16; ++jt) {
            v16bf bK;
            const __bf16* kp = &Ks[jt * 16 + l15][0];
            #pragma unroll
            for (int t = 0; t < 16; ++t)
                bK[t] = (t < 8) ? kp[half * 8 + t] : (__bf16)0.0f;   // K 16..31 zero pad
            v8f raw = (v8f){0.f, 0.f, 0.f, 0.f, 0.f, 0.f, 0.f, 0.f};
            raw = __builtin_amdgcn_wmma_f32_16x16x32_bf16(
                false, aQ, false, bK, (short)0, raw, false, false);
            #pragma unroll
            for (int e = 0; e < 8; ++e) {
                const float v = ((jt == strip) && (lane == e + half * 24))
                                ? -1e20f : raw[e];     // diagonal mask
                mx[e] = fmaxf(mx[e], v);
            }
        }
        #pragma unroll
        for (int e = 0; e < 8; ++e) {
            float v = mx[e];
            #pragma unroll
            for (int m = 1; m <= 8; m <<= 1) v = fmaxf(v, __shfl_xor(v, m, 32));
            mx[e] = v;
        }

        // ---- pass 2: exp + unnormalized P*V accumulation
        float lsum[8];
        #pragma unroll
        for (int e = 0; e < 8; ++e) lsum[e] = 0.f;
        v8f oacc[4];
        #pragma unroll
        for (int ct = 0; ct < 4; ++ct)
            oacc[ct] = (v8f){0.f, 0.f, 0.f, 0.f, 0.f, 0.f, 0.f, 0.f};

        for (int jj = 0; jj < 8; ++jj) {
            #pragma unroll
            for (int t = 0; t < 2; ++t) {
                const int jt = jj * 2 + t;
                v16bf bK;
                const __bf16* kp = &Ks[jt * 16 + l15][0];
                #pragma unroll
                for (int u = 0; u < 16; ++u)
                    bK[u] = (u < 8) ? kp[half * 8 + u] : (__bf16)0.0f;
                v8f raw = (v8f){0.f, 0.f, 0.f, 0.f, 0.f, 0.f, 0.f, 0.f};
                raw = __builtin_amdgcn_wmma_f32_16x16x32_bf16(
                    false, aQ, false, bK, (short)0, raw, false, false);
                #pragma unroll
                for (int e = 0; e < 8; ++e) {
                    const float v = ((jt == strip) && (lane == e + half * 24))
                                    ? -1e20f : raw[e];
                    const float p = __expf(v - mx[e]);
                    lsum[e] += p;
                    Ps[wave][e + half * 8][t * 16 + l15] = (__bf16)p;
                }
            }
            __builtin_amdgcn_wave_barrier();

            v16bf aP;
            {
                const __bf16* pp = &Ps[wave][l15][half * 16];
                #pragma unroll
                for (int t = 0; t < 8; ++t) {
                    aP[t]     = pp[t];
                    aP[8 + t] = pp[8 + t];
                }
            }
            const int jbase = jj * 32 + half * 16;
            #pragma unroll
            for (int ct = 0; ct < 4; ++ct) {
                const __bf16* vp = &Vt[ct * 16 + l15][jbase];
                v16bf bV;
                #pragma unroll
                for (int t = 0; t < 16; ++t) bV[t] = vp[t];
                oacc[ct] = __builtin_amdgcn_wmma_f32_16x16x32_bf16(
                    false, aP, false, bV, (short)0, oacc[ct], false, false);
            }
            __builtin_amdgcn_wave_barrier();
        }

        #pragma unroll
        for (int e = 0; e < 8; ++e) {
            float v = lsum[e];
            #pragma unroll
            for (int m = 1; m <= 8; m <<= 1) v += __shfl_xor(v, m, 32);
            lsum[e] = v;
        }

        // epilogue: normalize, add residual, scatter to NCHW
        #pragma unroll
        for (int ct = 0; ct < 4; ++ct) {
            const int c = ct * 16 + l15;
            #pragma unroll
            for (int e = 0; e < 8; ++e) {
                const int i = strip * 16 + e + half * 8;
                const int h = (i & 0xF0) + bh;
                const int w = ((i & 15) << 4) + bw;
                const size_t idx = (((size_t)n * C_ + c) * H_ + h) * W_ + w;
                out[idx] = oacc[ct][e] / lsum[e] + x[idx];
            }
        }
    }
}

// ---------------------------------------------------------------------------
extern "C" void kernel_launch(void* const* d_in, const int* in_sizes, int n_in,
                              void* d_out, int out_size, void* d_ws, size_t ws_size,
                              hipStream_t stream) {
    const float* x       = (const float*)d_in[0];
    const float* w_match = (const float*)d_in[1];
    const float* b_match = (const float*)d_in[2];
    const float* w_asm   = (const float*)d_in[3];
    const float* b_asm   = (const float*)d_in[4];
    float* out = (float*)d_out;

    float*  x_embed = (float*)((char*)d_ws + WS_XEMBED);   // 32 MiB
    __bf16* wrA     = (__bf16*)((char*)d_ws + WS_WRA);     // 18 KiB
    __bf16* wasmA   = (__bf16*)((char*)d_ws + WS_WASMA);   // 8 KiB

    pack_weights_kernel<<<1, 256, 0, stream>>>(w_match, w_asm, wrA, wasmA);
    conv3x3_wmma_kernel<<<dim3(4, 64, 8), 256, 0, stream>>>(x, wrA, b_match, x_embed);
    attn_kernel<<<dim3(N_ * NBLK), 256, 0, stream>>>(x_embed, x, wasmA, b_asm, out);
}
